// FakeNewsGAT_33595234189546
// MI455X (gfx1250) — matrix operand order repaired and checked by reading, not compiled, not run on hardware
//
#include <hip/hip_runtime.h>

typedef __attribute__((ext_vector_type(16))) _Float16 v16h;
typedef __attribute__((ext_vector_type(8)))  _Float16 v8h;
typedef __attribute__((ext_vector_type(8)))  float    v8f;

#define N_NODES  20000
#define N_EDGES  320000
#define IN_F     768
#define H1_HEADS 4
#define D1_F     128
#define HD1      512     // H1_HEADS * D1_F
#define D2_F     64
#define NCLS     2

// ---------- helpers ----------
static __device__ __forceinline__ unsigned fenc(float x) {
  unsigned u = __float_as_uint(x);
  return (u & 0x80000000u) ? ~u : (u | 0x80000000u);   // monotonic float->uint
}
static __device__ __forceinline__ float fdec(unsigned u) {
  unsigned v = (u & 0x80000000u) ? (u & 0x7fffffffu) : ~u;
  return __uint_as_float(v);
}
static __device__ __forceinline__ float elu1(float v) {
  return v > 0.f ? v : (expf(v) - 1.f);
}

// ---------- prep ----------
__global__ void k_fill_u32(unsigned* p, unsigned v, int n) {
  int i = blockIdx.x * blockDim.x + threadIdx.x;
  if (i < n) p[i] = v;
}
__global__ void k_cast_f16(const float* __restrict__ in, _Float16* __restrict__ out, int n) {
  int i = blockIdx.x * blockDim.x + threadIdx.x;
  if (i < n) out[i] = (_Float16)in[i];
}
// out[c*rows + r] = (f16) in[r*cols + c]   (row-major [rows x cols] -> transposed f16)
__global__ void k_transpose_f16(const float* __restrict__ in, _Float16* __restrict__ out,
                                int rows, int cols) {
  int i = blockIdx.x * blockDim.x + threadIdx.x;
  if (i >= rows * cols) return;
  int r = i / cols, c = i - r * cols;
  out[(size_t)c * rows + r] = (_Float16)in[i];
}

// ---------- WMMA GEMM: C[M x Nc] = A[M x K](f16,row-major) * B (Bt = [Nc x K] f16) ----------
// One wave computes a 16x64 output strip (4 tiles): the A fragment is loaded once per
// K-step and reused across 4 v_wmma_f32_16x16x32_f16, halving L2 traffic per FLOP.
// Fragments use contiguous k = 16*hi + j in A and B (wave32). Requires Ncols % 64 == 0, K % 32 == 0,
// M % 16 == 0 (20000 = 1250*16, 768/512 OK).
__global__ void k_wmma_gemm4(const _Float16* __restrict__ A, const _Float16* __restrict__ Bt,
                             float* __restrict__ C, int M, int K, int Ncols) {
  int wave = (int)((blockIdx.x * blockDim.x + threadIdx.x) >> 5);
  int lane = threadIdx.x & 31;
  int tilesN4 = Ncols >> 6;                  // groups of four 16-wide tiles
  int tilesM  = M >> 4;
  if (wave >= tilesM * tilesN4) return;
  int tm = wave / tilesN4;
  int tg = wave - tm * tilesN4;
  int l16 = lane & 15;
  int hi  = lane >> 4;                       // 0 for lanes 0-15, 1 for 16-31
  // A fragment: lane holds row (tm*16+l16); half j -> K = k0 + 16*hi + j
  const _Float16* arow = A  + (size_t)(tm * 16 + l16) * K + (hi ? 16 : 0);
  // B fragment (tile t): lane holds column (tg*64 + t*16 + l16); half j -> K = k0 + 16*hi + j
  const _Float16* bcol = Bt + (size_t)(tg * 64 + l16) * K + (hi ? 16 : 0);
  size_t bts = (size_t)16 * K;               // column-tile stride within Bt
  v8f acc0 = {}, acc1 = {}, acc2 = {}, acc3 = {};
  for (int k0 = 0; k0 < K; k0 += 32) {
    union { v16h v; v8h h[2]; } a, b0, b1, b2, b3;
    a.h[0]  = *(const v8h*)(arow + k0);
    a.h[1]  = *(const v8h*)(arow + k0 + 8);
    b0.h[0] = *(const v8h*)(bcol + k0);
    b0.h[1] = *(const v8h*)(bcol + k0 + 8);
    b1.h[0] = *(const v8h*)(bcol + bts + k0);
    b1.h[1] = *(const v8h*)(bcol + bts + k0 + 8);
    b2.h[0] = *(const v8h*)(bcol + 2 * bts + k0);
    b2.h[1] = *(const v8h*)(bcol + 2 * bts + k0 + 8);
    b3.h[0] = *(const v8h*)(bcol + 3 * bts + k0);
    b3.h[1] = *(const v8h*)(bcol + 3 * bts + k0 + 8);
    acc0 = __builtin_amdgcn_wmma_f32_16x16x32_f16(false, a.v, false, b0.v, (short)0, acc0, false, false);
    acc1 = __builtin_amdgcn_wmma_f32_16x16x32_f16(false, a.v, false, b1.v, (short)0, acc1, false, false);
    acc2 = __builtin_amdgcn_wmma_f32_16x16x32_f16(false, a.v, false, b2.v, (short)0, acc2, false, false);
    acc3 = __builtin_amdgcn_wmma_f32_16x16x32_f16(false, a.v, false, b3.v, (short)0, acc3, false, false);
  }
  // C/D layout: vgpr r -> row tm*16 + r + 8*hi, col = tile_base + l16
  float* crow = C + (size_t)(tm * 16 + hi * 8) * Ncols + tg * 64 + l16;
  #pragma unroll
  for (int r = 0; r < 8; ++r) {
    float* p = crow + (size_t)r * Ncols;
    p[0]  = acc0[r];
    p[16] = acc1[r];
    p[32] = acc2[r];
    p[48] = acc3[r];
  }
}

// ---------- attention scores: es/ed[n,h] = dot(H[n,h,:], a_{s,d}[h,:]) ----------
__global__ void k_scores(const float* __restrict__ Hm, const float* __restrict__ a_s,
                         const float* __restrict__ a_d, float* __restrict__ es,
                         float* __restrict__ ed, int n, int heads, int D) {
  int i = blockIdx.x * blockDim.x + threadIdx.x;
  if (i >= n * heads) return;
  int node = i / heads, h = i - node * heads;
  const float* hp = Hm + ((size_t)node * heads + h) * D;
  const float* as = a_s + (size_t)h * D;
  const float* ad = a_d + (size_t)h * D;
  float ss = 0.f, sd = 0.f;
  for (int d = 0; d < D; ++d) { float v = hp[d]; ss += v * as[d]; sd += v * ad[d]; }
  es[i] = ss; ed[i] = sd;
}

// ---------- softmax pass 1: segment max over dst ----------
__global__ void k_edge_max(const int* __restrict__ src, const int* __restrict__ dst,
                           const float* __restrict__ es, const float* __restrict__ ed,
                           unsigned* __restrict__ menc, int E, int n, int heads) {
  int i = blockIdx.x * blockDim.x + threadIdx.x;
  int tot = (E + n) * heads;
  if (i >= tot) return;
  int e = i / heads, h = i - e * heads;
  int s = (e < E) ? src[e] : (e - E);     // tail = self loops
  int d = (e < E) ? dst[e] : (e - E);
  float v = es[s * heads + h] + ed[d * heads + h];
  v = (v > 0.f) ? v : 0.2f * v;           // leaky_relu(0.2)
  atomicMax(&menc[d * heads + h], fenc(v));
}

// ---------- softmax pass 2: exp + segment sum; stash exp per edge ----------
__global__ void k_edge_den(const int* __restrict__ src, const int* __restrict__ dst,
                           const float* __restrict__ es, const float* __restrict__ ed,
                           const unsigned* __restrict__ menc, float* __restrict__ den,
                           float* __restrict__ exbuf, int E, int n, int heads) {
  int i = blockIdx.x * blockDim.x + threadIdx.x;
  int tot = (E + n) * heads;
  if (i >= tot) return;
  int e = i / heads, h = i - e * heads;
  int s = (e < E) ? src[e] : (e - E);
  int d = (e < E) ? dst[e] : (e - E);
  float v = es[s * heads + h] + ed[d * heads + h];
  v = (v > 0.f) ? v : 0.2f * v;
  float ex = expf(v - fdec(menc[d * heads + h]));
  exbuf[i] = ex;
  atomicAdd(&den[d * heads + h], ex);
}

// ---------- pass 3: out[dst] += alpha * H[src], 8 dims per thread ----------
__global__ void k_scatter(const int* __restrict__ src, const int* __restrict__ dst,
                          const float* __restrict__ exbuf, const float* __restrict__ den,
                          const float* __restrict__ Hm, float* __restrict__ out,
                          int E, int n, int heads, int D) {
  int chunks = D >> 3;
  int i = blockIdx.x * blockDim.x + threadIdx.x;
  int tot = (E + n) * heads * chunks;
  if (i >= tot) return;
  int c = i % chunks;
  int t = i / chunks;
  int h = t % heads;
  int e = t / heads;
  int s = (e < E) ? src[e] : (e - E);
  int d = (e < E) ? dst[e] : (e - E);
  float a = exbuf[(size_t)e * heads + h] / (den[d * heads + h] + 1e-16f);
  const float* hp = Hm  + ((size_t)s * heads + h) * D + c * 8;
  float*       op = out + ((size_t)d * heads + h) * D + c * 8;
  #pragma unroll
  for (int j = 0; j < 8; ++j) atomicAdd(&op[j], hp[j] * a);
}

// ---------- bias + ELU + recast to f16 (feeds next WMMA layer) ----------
__global__ void k_bias_elu_f16(const float* __restrict__ agg, const float* __restrict__ b,
                               _Float16* __restrict__ out, int n, int F) {
  int i = blockIdx.x * blockDim.x + threadIdx.x;
  if (i >= n * F) return;
  int f = i % F;
  out[i] = (_Float16)elu1(agg[i] + b[f]);
}

// ---------- head: out = elu(agg2 + b2) @ Wc + bc  (64 -> 2) ----------
__global__ void k_final(const float* __restrict__ agg2, const float* __restrict__ b2,
                        const float* __restrict__ Wc, const float* __restrict__ bc,
                        float* __restrict__ out, int n) {
  int i = blockIdx.x * blockDim.x + threadIdx.x;
  if (i >= n) return;
  const float* hp = agg2 + (size_t)i * D2_F;
  float o0 = bc[0], o1 = bc[1];
  for (int c = 0; c < D2_F; ++c) {
    float v = elu1(hp[c] + b2[c]);
    o0 += v * Wc[c * 2 + 0];
    o1 += v * Wc[c * 2 + 1];
  }
  out[(size_t)i * 2 + 0] = o0;
  out[(size_t)i * 2 + 1] = o1;
}

// ---------- launcher ----------
static inline int grid1(long long n, int bs = 256) { return (int)((n + bs - 1) / bs); }

extern "C" void kernel_launch(void* const* d_in, const int* in_sizes, int n_in,
                              void* d_out, int out_size, void* d_ws, size_t ws_size,
                              hipStream_t stream) {
  (void)in_sizes; (void)n_in; (void)out_size; (void)ws_size;
  const float* x   = (const float*)d_in[0];
  const int*   ei  = (const int*)d_in[1];
  const int*   src = ei;
  const int*   dst = ei + N_EDGES;
  const float* W1  = (const float*)d_in[2];
  const float* a1s = (const float*)d_in[3];
  const float* a1d = (const float*)d_in[4];
  const float* b1  = (const float*)d_in[5];
  const float* W2  = (const float*)d_in[6];
  const float* a2s = (const float*)d_in[7];
  const float* a2d = (const float*)d_in[8];
  const float* b2  = (const float*)d_in[9];
  const float* Wc  = (const float*)d_in[10];
  const float* bc  = (const float*)d_in[11];
  float* out = (float*)d_out;

  char* w = (char*)d_ws;
  size_t off = 0;
  auto take = [&](size_t bytes) { char* p = w + off; off += (bytes + 255) & ~(size_t)255; return p; };
  _Float16* bufA  = (_Float16*)take((size_t)N_NODES * IN_F * 2);  // x_f16, later h1_f16
  _Float16* w1t   = (_Float16*)take((size_t)HD1 * IN_F * 2);      // W1^T f16 [512 x 768]
  _Float16* w2t   = (_Float16*)take((size_t)D2_F * HD1 * 2);      // W2^T f16 [64 x 512]
  float*    Hbuf  = (float*)take((size_t)N_NODES * HD1 * 4);      // H1, later H2
  float*    es    = (float*)take((size_t)N_NODES * H1_HEADS * 4);
  float*    ed    = (float*)take((size_t)N_NODES * H1_HEADS * 4);
  unsigned* menc  = (unsigned*)take((size_t)N_NODES * H1_HEADS * 4);
  float*    den   = (float*)take((size_t)N_NODES * H1_HEADS * 4);
  float*    exbuf = (float*)take((size_t)(N_EDGES + N_NODES) * H1_HEADS * 4);
  float*    agg   = (float*)take((size_t)N_NODES * HD1 * 4);      // out1, later out2

  const int BS = 256;
  const int Etot = N_EDGES + N_NODES;

  // ===== layer 1 =====
  k_cast_f16<<<grid1((long long)N_NODES * IN_F), BS, 0, stream>>>(x, bufA, N_NODES * IN_F);
  k_transpose_f16<<<grid1((long long)IN_F * HD1), BS, 0, stream>>>(W1, w1t, IN_F, HD1);
  k_transpose_f16<<<grid1((long long)HD1 * D2_F), BS, 0, stream>>>(W2, w2t, HD1, D2_F);

  {
    long long waves = (long long)(N_NODES / 16) * (HD1 / 64);   // 1250 * 8 = 10000
    k_wmma_gemm4<<<grid1(waves * 32), BS, 0, stream>>>(bufA, w1t, Hbuf, N_NODES, IN_F, HD1);
  }
  k_scores<<<grid1((long long)N_NODES * H1_HEADS), BS, 0, stream>>>(
      Hbuf, a1s, a1d, es, ed, N_NODES, H1_HEADS, D1_F);

  k_fill_u32<<<grid1(N_NODES * H1_HEADS), BS, 0, stream>>>(menc, 0u, N_NODES * H1_HEADS);
  k_fill_u32<<<grid1(N_NODES * H1_HEADS), BS, 0, stream>>>((unsigned*)den, 0u, N_NODES * H1_HEADS);
  k_fill_u32<<<grid1((long long)N_NODES * HD1), BS, 0, stream>>>((unsigned*)agg, 0u, N_NODES * HD1);

  k_edge_max<<<grid1((long long)Etot * H1_HEADS), BS, 0, stream>>>(
      src, dst, es, ed, menc, N_EDGES, N_NODES, H1_HEADS);
  k_edge_den<<<grid1((long long)Etot * H1_HEADS), BS, 0, stream>>>(
      src, dst, es, ed, menc, den, exbuf, N_EDGES, N_NODES, H1_HEADS);
  k_scatter<<<grid1((long long)Etot * H1_HEADS * (D1_F / 8)), BS, 0, stream>>>(
      src, dst, exbuf, den, Hbuf, agg, N_EDGES, N_NODES, H1_HEADS, D1_F);

  // bias + ELU -> f16 (reuse bufA as h1_f16)
  k_bias_elu_f16<<<grid1((long long)N_NODES * HD1), BS, 0, stream>>>(agg, b1, bufA, N_NODES, HD1);

  // ===== layer 2 =====
  {
    long long waves = (long long)(N_NODES / 16) * (D2_F / 64);  // 1250
    k_wmma_gemm4<<<grid1(waves * 32), BS, 0, stream>>>(bufA, w2t, Hbuf, N_NODES, HD1, D2_F);
  }
  k_scores<<<grid1((long long)N_NODES), BS, 0, stream>>>(Hbuf, a2s, a2d, es, ed, N_NODES, 1, D2_F);

  k_fill_u32<<<grid1(N_NODES), BS, 0, stream>>>(menc, 0u, N_NODES);
  k_fill_u32<<<grid1(N_NODES), BS, 0, stream>>>((unsigned*)den, 0u, N_NODES);
  k_fill_u32<<<grid1((long long)N_NODES * D2_F), BS, 0, stream>>>((unsigned*)agg, 0u, N_NODES * D2_F);

  k_edge_max<<<grid1((long long)Etot), BS, 0, stream>>>(src, dst, es, ed, menc, N_EDGES, N_NODES, 1);
  k_edge_den<<<grid1((long long)Etot), BS, 0, stream>>>(src, dst, es, ed, menc, den, exbuf,
                                                        N_EDGES, N_NODES, 1);
  k_scatter<<<grid1((long long)Etot * (D2_F / 8)), BS, 0, stream>>>(
      src, dst, exbuf, den, Hbuf, agg, N_EDGES, N_NODES, 1, D2_F);

  // ===== classifier head =====
  k_final<<<grid1(N_NODES), BS, 0, stream>>>(agg, b2, Wc, bc, out, N_NODES);
}
